// WavLMEncoder_50903952392465
// MI455X (gfx1250) — matrix operand order repaired and checked by reading, not compiled, not run on hardware
//
#include <hip/hip_runtime.h>
#include <cstdint>

// ---------------------------------------------------------------------------
// Types / constants
// ---------------------------------------------------------------------------
typedef __bf16 bf16_t;
typedef __bf16 v16bf __attribute__((ext_vector_type(16)));
typedef float  v8f   __attribute__((ext_vector_type(8)));

constexpr int cB = 8, cD = 768, cT = 1024, cH = 8, cDH = 96, cFF = 3072;
constexpr int cBT   = cB * cT;    // 8192
constexpr int cQKV  = 3 * cD;     // 2304
constexpr int cCONVK = 31;
constexpr int cHALF = 160, cSMALL = 80;

union Frag {
  v16bf v;
  uint4 q[2];
};

enum { EPI_PLAIN = 0, EPI_GELU = 1, EPI_RESID = 2, EPI_RESID_HALF = 3 };

// ---------------------------------------------------------------------------
// Tiled bf16 WMMA GEMM:  out = epi(A[M,K] @ B[K,N] + bias, resid)
// Block tile 128x128x32, 8 waves, wave tile 32x64 (2x4 of 16x16 WMMA tiles).
// ---------------------------------------------------------------------------
constexpr int GBM = 128, GBN = 128, GBK = 32;

template <int EPI>
__global__ __launch_bounds__(256)
void k_gemm(const bf16_t* __restrict__ A, const bf16_t* __restrict__ Bm,
            const float* __restrict__ bias, const float* __restrict__ resid,
            float* __restrict__ outf, bf16_t* __restrict__ outb,
            int M, int N, int K)
{
  __shared__ bf16_t As[GBM][GBK + 8];   // row-major [m][k]
  __shared__ bf16_t Bs[GBN][GBK + 8];   // transposed [n][k]

  const int tid  = threadIdx.x;
  const int lane = tid & 31;
  const int wave = tid >> 5;
  const int wr   = wave & 3;       // row-wave: rows wr*32
  const int wc   = wave >> 2;      // col-wave: cols wc*64
  const int half = lane >> 4;      // 0|1 (lane group)
  const int l16  = lane & 15;
  const int bm   = blockIdx.y * GBM;
  const int bn   = blockIdx.x * GBN;

  v8f acc[2][4];
#pragma unroll
  for (int i = 0; i < 2; ++i)
#pragma unroll
    for (int j = 0; j < 4; ++j)
#pragma unroll
      for (int r = 0; r < 8; ++r) acc[i][j][r] = 0.0f;

  for (int k0 = 0; k0 < K; k0 += GBK) {
    // stage A tile (128x32): 512 chunks of 8 bf16 (16B)
    for (int s = tid; s < GBM * GBK / 8; s += 256) {
      const int row = s >> 2;
      const int cc  = (s & 3) * 8;
      uint4 vq = *reinterpret_cast<const uint4*>(A + (size_t)(bm + row) * K + k0 + cc);
      *reinterpret_cast<uint4*>(&As[row][cc]) = vq;
    }
    // stage B tile (32x128) transposed into Bs[n][k]
    for (int s = tid; s < GBK * GBN / 8; s += 256) {
      const int kk = s >> 4;
      const int cc = (s & 15) * 8;
      uint4 vq = *reinterpret_cast<const uint4*>(Bm + (size_t)(k0 + kk) * N + bn + cc);
      const bf16_t* e = reinterpret_cast<const bf16_t*>(&vq);
#pragma unroll
      for (int t = 0; t < 8; ++t) Bs[cc + t][kk] = e[t];
    }
    if (k0 + GBK < K) {   // gfx1250 global_prefetch_b8 for next K tile
      __builtin_prefetch(A + (size_t)(bm + (tid & 127)) * K + (k0 + GBK), 0, 1);
      __builtin_prefetch(Bm + (size_t)(k0 + GBK + (tid & 31)) * N + bn + ((tid >> 5) * 16), 0, 1);
    }
    __syncthreads();

    // A frag (16x32 bf16): lanes<16 -> K{0..7,16..23}, lanes>=16 -> K{8..15,24..31}
    Frag af[2], bf[4];
#pragma unroll
    for (int i = 0; i < 2; ++i) {
      const bf16_t* p = &As[wr * 32 + i * 16 + l16][0];
      af[i].q[0] = *reinterpret_cast<const uint4*>(p + half * 8);
      af[i].q[1] = *reinterpret_cast<const uint4*>(p + 16 + half * 8);
    }
    // B frag (32x16): lanes<16 -> K{0..7,16..23}, lanes>=16 -> K{8..15,24..31}, col = lane%16
#pragma unroll
    for (int j = 0; j < 4; ++j) {
      const bf16_t* p = &Bs[wc * 64 + j * 16 + l16][0];
      bf[j].q[0] = *reinterpret_cast<const uint4*>(p + half * 8);
      bf[j].q[1] = *reinterpret_cast<const uint4*>(p + 16 + half * 8);
    }
#pragma unroll
    for (int i = 0; i < 2; ++i)
#pragma unroll
      for (int j = 0; j < 4; ++j)
        acc[i][j] = __builtin_amdgcn_wmma_f32_16x16x32_bf16(
            false, af[i].v, false, bf[j].v, (short)0, acc[i][j], false, false);
    __syncthreads();
  }

  // epilogue: C layout -> row = r + 8*half, col = lane%16
#pragma unroll
  for (int i = 0; i < 2; ++i)
#pragma unroll
    for (int j = 0; j < 4; ++j) {
      const int rowb = bm + wr * 32 + i * 16 + half * 8;
      const int col  = bn + wc * 64 + j * 16 + l16;
      const float bv = bias ? bias[col] : 0.0f;
#pragma unroll
      for (int r = 0; r < 8; ++r) {
        float a = acc[i][j][r] + bv;
        const size_t idx = (size_t)(rowb + r) * N + col;
        float o;
        if constexpr (EPI == EPI_GELU)            o = 0.5f * a * (1.0f + erff(a * 0.70710678f));
        else if constexpr (EPI == EPI_RESID)      o = resid[idx] + a;
        else if constexpr (EPI == EPI_RESID_HALF) o = resid[idx] + 0.5f * a;
        else                                      o = a;
        if (outf) outf[idx] = o;
        if (outb) outb[idx] = (bf16_t)o;
      }
    }
}

// ---------------------------------------------------------------------------
// Gated relative position bias (computed per score element)
// ---------------------------------------------------------------------------
__device__ inline float rel_bias_val(int tq, int ts, int h, const float* __restrict__ rel,
                                     float gu, float gr, float gs)
{
  int rp = ts - tq;
  int sign = (rp >= 0) ? 1 : 0;
  int ra = rp < 0 ? -rp : rp;
  int bkt;
  if (ra < cSMALL) {
    bkt = ra;
  } else {
    // log(ra/80)/log(800/80) * 80, trunc toward zero, clamp to 159
    float lr = __logf((float)ra * (1.0f / (float)cSMALL)) * 0.4342944819f;
    int lp = cSMALL + (int)(lr * (float)(cHALF - cSMALL));
    bkt = lp < (cHALF - 1) ? lp : (cHALF - 1);
  }
  bkt += sign * cHALF;
  float dtab = rel[bkt * cH + h];
  return dtab * (1.0f + gu) + (1.0f - gu) * gs * gr * dtab;
}

// ---------------------------------------------------------------------------
// Flash attention: one wave per (b, h, 16-row tile). Online softmax.
// Score tiles via WMMA (Q 16x96 . K^T), O accum via WMMA (P 16x32 . V 32x96).
// ---------------------------------------------------------------------------
__global__ __launch_bounds__(32)
void k_attn(const bf16_t* __restrict__ Qg, const bf16_t* __restrict__ Kg,
            const bf16_t* __restrict__ Vg,
            const float* __restrict__ gu, const float* __restrict__ gr,
            const float* __restrict__ rel, const float* __restrict__ gscale,
            bf16_t* __restrict__ out)
{
  __shared__ bf16_t Pl[16][40];   // P tile 16x32 (A-layout round trip)
  __shared__ bf16_t Vt[96][40];   // V tile transposed [n][k], k=0..31

  const int lane = threadIdx.x;
  const int half = lane >> 4;
  const int l16  = lane & 15;
  const int rt   = blockIdx.x;    // 16-row tile index
  const int h    = blockIdx.y;
  const int b    = blockIdx.z;

  const size_t headoff = ((size_t)(b * cH + h)) * cT * cDH;
  const bf16_t* Q = Qg + headoff;
  const bf16_t* K = Kg + headoff;
  const bf16_t* V = Vg + headoff;
  const float*  GU = gu + (size_t)(b * cH + h) * cT;
  const float*  GR = gr + (size_t)(b * cH + h) * cT;
  const float   gs = gscale[h];

  // Q fragments for the 3 K-slices of DH=96
  Frag qf[3];
  {
    const int tqa = rt * 16 + l16;
#pragma unroll
    for (int c = 0; c < 3; ++c) {
      const bf16_t* p = Q + (size_t)tqa * cDH + c * 32;
      qf[c].q[0] = *reinterpret_cast<const uint4*>(p + half * 8);
      qf[c].q[1] = *reinterpret_cast<const uint4*>(p + 16 + half * 8);
    }
  }

  float m[8], l[8], guv[8], grv[8];
  v8f o[6];
#pragma unroll
  for (int r = 0; r < 8; ++r) {
    m[r] = -1e30f; l[r] = 0.0f;
    const int tq = rt * 16 + r + half * 8;
    guv[r] = GU[tq]; grv[r] = GR[tq];
  }
#pragma unroll
  for (int t = 0; t < 6; ++t)
#pragma unroll
    for (int r = 0; r < 8; ++r) o[t][r] = 0.0f;

  const float inv = 0.1020620726f;  // 1/sqrt(96)

  for (int j = 0; j < cT; j += 32) {
    // ---- scores: two 16x16 tiles, accumulate over DH (3 x K32) ----
    v8f s0, s1;
#pragma unroll
    for (int r = 0; r < 8; ++r) { s0[r] = 0.0f; s1[r] = 0.0f; }
#pragma unroll
    for (int c = 0; c < 3; ++c) {
      Frag kb0, kb1;
      const bf16_t* p0 = K + (size_t)(j + l16) * cDH + c * 32;
      kb0.q[0] = *reinterpret_cast<const uint4*>(p0 + half * 8);
      kb0.q[1] = *reinterpret_cast<const uint4*>(p0 + 16 + half * 8);
      const bf16_t* p1 = K + (size_t)(j + 16 + l16) * cDH + c * 32;
      kb1.q[0] = *reinterpret_cast<const uint4*>(p1 + half * 8);
      kb1.q[1] = *reinterpret_cast<const uint4*>(p1 + 16 + half * 8);
      s0 = __builtin_amdgcn_wmma_f32_16x16x32_bf16(false, qf[c].v, false, kb0.v, (short)0, s0, false, false);
      s1 = __builtin_amdgcn_wmma_f32_16x16x32_bf16(false, qf[c].v, false, kb1.v, (short)0, s1, false, false);
    }

    // ---- bias + online softmax (row reductions across 16-lane group) ----
    float a0[8], a1[8], mnew[8];
#pragma unroll
    for (int r = 0; r < 8; ++r) {
      const int tq = rt * 16 + r + half * 8;
      a0[r] = s0[r] * inv + rel_bias_val(tq, j + l16,      h, rel, guv[r], grv[r], gs);
      a1[r] = s1[r] * inv + rel_bias_val(tq, j + 16 + l16, h, rel, guv[r], grv[r], gs);
      float mm = fmaxf(a0[r], a1[r]);
#pragma unroll
      for (int off = 1; off < 16; off <<= 1) mm = fmaxf(mm, __shfl_xor(mm, off, 32));
      mnew[r] = fmaxf(m[r], mm);
    }
#pragma unroll
    for (int r = 0; r < 8; ++r) {
      const float sc = __expf(m[r] - mnew[r]);
      m[r] = mnew[r];
      const float p0 = __expf(a0[r] - mnew[r]);
      const float p1 = __expf(a1[r] - mnew[r]);
      float rs = p0 + p1;
#pragma unroll
      for (int off = 1; off < 16; off <<= 1) rs += __shfl_xor(rs, off, 32);
      l[r] = l[r] * sc + rs;
#pragma unroll
      for (int t = 0; t < 6; ++t) o[t][r] *= sc;
      Pl[r + half * 8][l16]      = (bf16_t)p0;   // C-layout -> LDS
      Pl[r + half * 8][16 + l16] = (bf16_t)p1;
    }

    // ---- stage V tile (32 x 96) transposed into Vt[n][k] ----
    for (int s = lane; s < 32 * 12; s += 32) {
      const int kk = s & 31;
      const int ch = s >> 5;
      uint4 vq = *reinterpret_cast<const uint4*>(V + (size_t)(j + kk) * cDH + ch * 8);
      const bf16_t* e = reinterpret_cast<const bf16_t*>(&vq);
#pragma unroll
      for (int t = 0; t < 8; ++t) Vt[ch * 8 + t][kk] = e[t];
    }
    __syncthreads();

    // ---- O += P (16x32) @ V (32x96): 6 WMMAs ----
    Frag pf;
    {
      const bf16_t* p = &Pl[l16][0];
      pf.q[0] = *reinterpret_cast<const uint4*>(p + half * 8);
      pf.q[1] = *reinterpret_cast<const uint4*>(p + 16 + half * 8);
    }
#pragma unroll
    for (int t = 0; t < 6; ++t) {
      Frag vf;
      const bf16_t* p = &Vt[t * 16 + l16][0];
      vf.q[0] = *reinterpret_cast<const uint4*>(p + half * 8);
      vf.q[1] = *reinterpret_cast<const uint4*>(p + 16 + half * 8);
      o[t] = __builtin_amdgcn_wmma_f32_16x16x32_bf16(false, pf.v, false, vf.v, (short)0, o[t], false, false);
    }
    __syncthreads();
  }

  // normalize and write attn output as [B*T, D] bf16, col = h*96 + n
#pragma unroll
  for (int t = 0; t < 6; ++t)
#pragma unroll
    for (int r = 0; r < 8; ++r) {
      const int tq = rt * 16 + r + half * 8;
      const float ov = o[t][r] / l[r];
      out[((size_t)(b * cT + tq)) * cD + h * cDH + t * 16 + l16] = (bf16_t)ov;
    }
}

// ---------------------------------------------------------------------------
// Elementwise / reduction helpers
// ---------------------------------------------------------------------------
__global__ void k_cvt(const float* __restrict__ a, bf16_t* __restrict__ o, int n) {
  int i = blockIdx.x * blockDim.x + threadIdx.x;
  if (i < n) o[i] = (bf16_t)a[i];
}

__global__ void k_cvt_T(const float* __restrict__ a, bf16_t* __restrict__ o, int R, int C) {
  int i = blockIdx.x * blockDim.x + threadIdx.x;   // a[R,C] -> o[C,R]
  if (i < R * C) {
    int r = i / C, c = i % C;
    o[(size_t)c * R + r] = (bf16_t)a[i];
  }
}

__global__ void k_transpose_in(const float* __restrict__ x, float* __restrict__ sf,
                               bf16_t* __restrict__ sb) {
  int i = blockIdx.x * blockDim.x + threadIdx.x;   // dest [B,T,D]
  if (i >= cB * cT * cD) return;
  int d = i % cD;
  int t = (i / cD) % cT;
  int b = i / (cD * cT);
  float v = x[((size_t)b * cD + d) * cT + t];
  sf[i] = v; sb[i] = (bf16_t)v;
}

__global__ void k_transpose_out(const float* __restrict__ s, float* __restrict__ y) {
  int i = blockIdx.x * blockDim.x + threadIdx.x;   // dest [B,D,T]
  if (i >= cB * cD * cT) return;
  int t = i % cT;
  int d = (i / cT) % cD;
  int b = i / (cD * cT);
  y[i] = s[((size_t)b * cT + t) * cD + d];
}

__global__ void k_split_qkv(const float* __restrict__ qkv, bf16_t* __restrict__ q,
                            bf16_t* __restrict__ k, bf16_t* __restrict__ v) {
  int i = blockIdx.x * blockDim.x + threadIdx.x;   // [B,H,T,DH]
  if (i >= cB * cH * cT * cDH) return;
  int d = i % cDH;
  int t = (i / cDH) % cT;
  int h = (i / (cDH * cT)) % cH;
  int b = i / (cDH * cT * cH);
  size_t src = ((size_t)(b * cT + t)) * cQKV + h * cDH + d;
  q[i] = (bf16_t)qkv[src];
  k[i] = (bf16_t)qkv[src + cD];
  v[i] = (bf16_t)qkv[src + 2 * cD];
}

__global__ void k_gates(const float* __restrict__ qkv, const float* __restrict__ wu,
                        const float* __restrict__ wr, float* __restrict__ gu,
                        float* __restrict__ gr) {
  int i = blockIdx.x * blockDim.x + threadIdx.x;   // [B,H,T]
  if (i >= cB * cH * cT) return;
  int t = i % cT;
  int h = (i / cT) % cH;
  int b = i / (cT * cH);
  const float* qp = qkv + ((size_t)(b * cT + t)) * cQKV + h * cDH;
  float su = 0.f, sr = 0.f;
  for (int d = 0; d < cDH; ++d) {
    su += qp[d] * wu[h * cDH + d];
    sr += qp[d] * wr[h * cDH + d];
  }
  gu[i] = 1.0f / (1.0f + __expf(-su));
  gr[i] = 1.0f / (1.0f + __expf(-sr));
}

__global__ __launch_bounds__(256)
void k_gn_stats(const float* __restrict__ x, float* __restrict__ stats) {
  __shared__ float ss[256], sq[256];
  const int b = blockIdx.x, tid = threadIdx.x;
  const float* p = x + (size_t)b * cD * cT;
  float s = 0.f, q = 0.f;
  for (int i = tid; i < cD * cT; i += 256) { float v = p[i]; s += v; q += v * v; }
  ss[tid] = s; sq[tid] = q;
  __syncthreads();
  for (int o2 = 128; o2 > 0; o2 >>= 1) {
    if (tid < o2) { ss[tid] += ss[tid + o2]; sq[tid] += sq[tid + o2]; }
    __syncthreads();
  }
  if (tid == 0) {
    const float n = (float)(cD * cT);
    float mean = ss[0] / n;
    float var = sq[0] / n - mean * mean;
    stats[b * 2] = mean;
    stats[b * 2 + 1] = rsqrtf(var + 1e-5f);
  }
}

template <int SILU>
__global__ void k_gn_apply(const float* __restrict__ x, const float* __restrict__ stats,
                           const float* __restrict__ w, const float* __restrict__ bb,
                           bf16_t* __restrict__ out) {
  int i = blockIdx.x * blockDim.x + threadIdx.x;   // [B,T,D]
  if (i >= cB * cT * cD) return;
  int d = i % cD;
  int b = i / (cT * cD);
  float y = (x[i] - stats[b * 2]) * stats[b * 2 + 1] * w[d] + bb[d];
  if constexpr (SILU) y = y / (1.0f + __expf(-y));
  out[i] = (bf16_t)y;
}

__global__ void k_glu(const float* __restrict__ pw1, float* __restrict__ out) {
  int i = blockIdx.x * blockDim.x + threadIdx.x;   // [BT,D]
  if (i >= cBT * cD) return;
  int bt = i / cD, d = i % cD;
  float a = pw1[(size_t)bt * (2 * cD) + d];
  float g = pw1[(size_t)bt * (2 * cD) + cD + d];
  out[i] = a * (1.0f / (1.0f + __expf(-g)));
}

__global__ void k_dwconv(const float* __restrict__ xin, const float* __restrict__ w,
                         const float* __restrict__ bb, float* __restrict__ out) {
  int i = blockIdx.x * blockDim.x + threadIdx.x;   // [B,T,D]
  if (i >= cB * cT * cD) return;
  int d = i % cD;
  int t = (i / cD) % cT;
  int b = i / (cD * cT);
  float s = bb[d];
#pragma unroll
  for (int kk = 0; kk < cCONVK; ++kk) {
    int tt = t + kk - cCONVK / 2;
    if (tt >= 0 && tt < cT) s += xin[((size_t)(b * cT + tt)) * cD + d] * w[d * cCONVK + kk];
  }
  out[i] = s;
}

// ---------------------------------------------------------------------------
// Host launcher
// ---------------------------------------------------------------------------
extern "C" void kernel_launch(void* const* d_in, const int* in_sizes, int n_in,
                              void* d_out, int out_size, void* d_ws, size_t ws_size,
                              hipStream_t stream) {
  (void)in_sizes; (void)n_in; (void)out_size; (void)ws_size;
  const float* x      = (const float*)d_in[0];
  const float* w_ff1a = (const float*)d_in[1];
  const float* b_ff1a = (const float*)d_in[2];
  const float* w_ff1b = (const float*)d_in[3];
  const float* b_ff1b = (const float*)d_in[4];
  const float* w_qkv  = (const float*)d_in[5];
  const float* b_qkv  = (const float*)d_in[6];
  const float* w_out  = (const float*)d_in[7];
  const float* b_out  = (const float*)d_in[8];
  const float* rel    = (const float*)d_in[9];
  const float* gate_u = (const float*)d_in[10];
  const float* gate_w = (const float*)d_in[11];
  const float* gate_s = (const float*)d_in[12];
  const float* gn1_w  = (const float*)d_in[13];
  const float* gn1_b  = (const float*)d_in[14];
  const float* pw1_w  = (const float*)d_in[15];
  const float* pw1_b  = (const float*)d_in[16];
  const float* dw_w   = (const float*)d_in[17];
  const float* dw_b   = (const float*)d_in[18];
  const float* gn2_w  = (const float*)d_in[19];
  const float* gn2_b  = (const float*)d_in[20];
  const float* pw2_w  = (const float*)d_in[21];
  const float* pw2_b  = (const float*)d_in[22];
  const float* w_ff2a = (const float*)d_in[23];
  const float* b_ff2a = (const float*)d_in[24];
  const float* w_ff2b = (const float*)d_in[25];
  const float* b_ff2b = (const float*)d_in[26];

  char* base = (char*)d_ws;
  size_t cur = 0;
  auto alloc = [&](size_t bytes) -> void* {
    cur = (cur + 255) & ~(size_t)255;
    void* p = base + cur;
    cur += bytes;
    return p;
  };

  float*  s_f32   = (float*) alloc((size_t)cBT * cD * 4);    // also glu / final s
  bf16_t* s_bf    = (bf16_t*)alloc((size_t)cBT * cD * 2);    // also gn normalized
  float*  s2_f32  = (float*) alloc((size_t)cBT * cD * 4);    // also dwconv out
  bf16_t* s2_bf   = (bf16_t*)alloc((size_t)cBT * cD * 2);    // also attn out
  float*  s3_f32  = (float*) alloc((size_t)cBT * cD * 4);
  bf16_t* s3_bf   = (bf16_t*)alloc((size_t)cBT * cD * 2);
  float*  c_f32   = (float*) alloc((size_t)cBT * cD * 4);
  bf16_t* c_bf    = (bf16_t*)alloc((size_t)cBT * cD * 2);
  bf16_t* hff_bf  = (bf16_t*)alloc((size_t)cBT * cFF * 2);   // FF hidden (reused FF2)
  float*  qkv_f32 = (float*) alloc((size_t)cBT * cQKV * 4);  // also pw1 out
  bf16_t* q_bf    = (bf16_t*)alloc((size_t)cB * cH * cT * cDH * 2);
  bf16_t* k_bf    = (bf16_t*)alloc((size_t)cB * cH * cT * cDH * 2);
  bf16_t* v_bf    = (bf16_t*)alloc((size_t)cB * cH * cT * cDH * 2);
  float*  gu_f    = (float*) alloc((size_t)cB * cH * cT * 4);
  float*  gr_f    = (float*) alloc((size_t)cB * cH * cT * 4);
  float*  stats1  = (float*) alloc(cB * 2 * 4);
  float*  stats2  = (float*) alloc(cB * 2 * 4);
  bf16_t* wff1a_bf = (bf16_t*)alloc((size_t)cD * cFF * 2);
  bf16_t* wff1b_bf = (bf16_t*)alloc((size_t)cFF * cD * 2);
  bf16_t* wqkv_bf  = (bf16_t*)alloc((size_t)cD * cQKV * 2);
  bf16_t* wout_bf  = (bf16_t*)alloc((size_t)cD * cD * 2);
  bf16_t* pw1T_bf  = (bf16_t*)alloc((size_t)cD * 2 * cD * 2);
  bf16_t* pw2T_bf  = (bf16_t*)alloc((size_t)cD * cD * 2);
  bf16_t* wff2a_bf = (bf16_t*)alloc((size_t)cD * cFF * 2);
  bf16_t* wff2b_bf = (bf16_t*)alloc((size_t)cFF * cD * 2);

  // lifetime-based aliases (non-overlapping uses)
  bf16_t* gn_bf    = s_bf;      // gn1/gn2 normalized activations (bf16)
  float*  conv_f32 = s2_f32;    // depthwise-conv output
  bf16_t* attn_bf  = s2_bf;     // attention output
  float*  pw1_f32  = qkv_f32;   // pointwise-conv-1 output
  float*  glu_f32  = s_f32;     // GLU output
  float*  sfin     = s_f32;     // final s (pre-transpose)

  const int TPB = 256;
  auto g1 = [&](int n) { return dim3((unsigned)((n + TPB - 1) / TPB)); };
  const dim3 blk(TPB);

  // weight conversion (bf16; pw1/pw2 transposed to [K,N])
  k_cvt<<<g1(cD * cFF), blk, 0, stream>>>(w_ff1a, wff1a_bf, cD * cFF);
  k_cvt<<<g1(cFF * cD), blk, 0, stream>>>(w_ff1b, wff1b_bf, cFF * cD);
  k_cvt<<<g1(cD * cQKV), blk, 0, stream>>>(w_qkv, wqkv_bf, cD * cQKV);
  k_cvt<<<g1(cD * cD), blk, 0, stream>>>(w_out, wout_bf, cD * cD);
  k_cvt_T<<<g1(2 * cD * cD), blk, 0, stream>>>(pw1_w, pw1T_bf, 2 * cD, cD);
  k_cvt_T<<<g1(cD * cD), blk, 0, stream>>>(pw2_w, pw2T_bf, cD, cD);
  k_cvt<<<g1(cD * cFF), blk, 0, stream>>>(w_ff2a, wff2a_bf, cD * cFF);
  k_cvt<<<g1(cFF * cD), blk, 0, stream>>>(w_ff2b, wff2b_bf, cFF * cD);

  // 1. [B,D,T] -> s [B,T,D] (f32 + bf16)
  k_transpose_in<<<g1(cBT * cD), blk, 0, stream>>>(x, s_f32, s_bf);

  // 2-3. FF1: gelu(s@Wa+ba) ; s2 = s + 0.5*(h@Wb+bb)
  k_gemm<EPI_GELU><<<dim3(cFF / GBN, cBT / GBM), blk, 0, stream>>>(
      s_bf, wff1a_bf, b_ff1a, nullptr, nullptr, hff_bf, cBT, cFF, cD);
  k_gemm<EPI_RESID_HALF><<<dim3(cD / GBN, cBT / GBM), blk, 0, stream>>>(
      hff_bf, wff1b_bf, b_ff1b, s_f32, s2_f32, s2_bf, cBT, cD, cFF);

  // 4-6. QKV projection, split to [B,H,T,DH] bf16, sigmoid gates
  k_gemm<EPI_PLAIN><<<dim3(cQKV / GBN, cBT / GBM), blk, 0, stream>>>(
      s2_bf, wqkv_bf, b_qkv, nullptr, qkv_f32, nullptr, cBT, cQKV, cD);
  k_split_qkv<<<g1(cB * cH * cT * cDH), blk, 0, stream>>>(qkv_f32, q_bf, k_bf, v_bf);
  k_gates<<<g1(cB * cH * cT), blk, 0, stream>>>(qkv_f32, gate_u, gate_w, gu_f, gr_f);

  // 7. flash attention with gated relative position bias
  k_attn<<<dim3(cT / 16, cH, cB), dim3(32), 0, stream>>>(
      q_bf, k_bf, v_bf, gu_f, gr_f, rel, gate_s, attn_bf);

  // 8. s3 = s2 + attn@Wout + bout
  k_gemm<EPI_RESID><<<dim3(cD / GBN, cBT / GBM), blk, 0, stream>>>(
      attn_bf, wout_bf, b_out, s2_f32, s3_f32, s3_bf, cBT, cD, cD);

  // 9-16. conv module
  k_gn_stats<<<dim3(cB), dim3(256), 0, stream>>>(s3_f32, stats1);
  k_gn_apply<0><<<g1(cBT * cD), blk, 0, stream>>>(s3_f32, stats1, gn1_w, gn1_b, gn_bf);
  k_gemm<EPI_PLAIN><<<dim3(2 * cD / GBN, cBT / GBM), blk, 0, stream>>>(
      gn_bf, pw1T_bf, pw1_b, nullptr, pw1_f32, nullptr, cBT, 2 * cD, cD);
  k_glu<<<g1(cBT * cD), blk, 0, stream>>>(pw1_f32, glu_f32);
  k_dwconv<<<g1(cBT * cD), blk, 0, stream>>>(glu_f32, dw_w, dw_b, conv_f32);
  k_gn_stats<<<dim3(cB), dim3(256), 0, stream>>>(conv_f32, stats2);
  k_gn_apply<1><<<g1(cBT * cD), blk, 0, stream>>>(conv_f32, stats2, gn2_w, gn2_b, gn_bf);
  k_gemm<EPI_RESID><<<dim3(cD / GBN, cBT / GBM), blk, 0, stream>>>(
      gn_bf, pw2T_bf, pw2_b, s3_f32, c_f32, c_bf, cBT, cD, cD);

  // 17-18. FF2
  k_gemm<EPI_GELU><<<dim3(cFF / GBN, cBT / GBM), blk, 0, stream>>>(
      c_bf, wff2a_bf, b_ff2a, nullptr, nullptr, hff_bf, cBT, cFF, cD);
  k_gemm<EPI_RESID_HALF><<<dim3(cD / GBN, cBT / GBM), blk, 0, stream>>>(
      hff_bf, wff2b_bf, b_ff2b, c_f32, sfin, nullptr, cBT, cD, cFF);

  // 19. [B,T,D] -> [B,D,T] output (f32)
  k_transpose_out<<<g1(cB * cD * cT), blk, 0, stream>>>(sfin, (float*)d_out);
}
